// CompleteSpatialConvolution_41747082117332
// MI455X (gfx1250) — matrix operand order repaired and checked by reading, not compiled, not run on hardware
//
#include <hip/hip_runtime.h>
#include <hip/hip_bf16.h>
#include <math.h>

// Problem constants (from reference)
#define NN    256
#define F0V   128
#define F1V   64
#define RBV   32
#define KSEQV 16
#define EMBV  32
#define CINV  321   // F0 + F1 + 1 + F0
#define NIV   192   // F0 + F1
#define OUTD  323   // F0 + 3*F1 + 3
#define FROW  320   // feat row stride = F0 + 3*F1

// Workspace layout (float offsets)
#define WS_P0   0         // [256][128]
#define WS_Q0   32768     // [256][128]
#define WS_V0   65536     // [256][3][128]
#define WS_P0W  163840    // [256][32]
#define WS_Q0W  172032    // [256][32]
#define WS_V0W  180224    // [256][3][32]
#define WS_SA   204800    // [256][32]
#define WS_A1   212992    // [256][3][64]
#define WS_S1   262144    // [256][64]
#define WS_B1   278528    // [256][3][64]
#define WS_E1   327680    // [256][3][64]
#define WS_PE   376832    // [33][32]

typedef __attribute__((ext_vector_type(16))) _Float16 v16h;
typedef __attribute__((ext_vector_type(8)))  float    v8f;

__device__ __forceinline__ float siluf(float x) { return x / (1.0f + __expf(-x)); }

// 16-bit A-matrix 16x32 lane layout (ISA 7.12.2): element (m,k) lives at
// lane = (m&15) + 16*((k>>3)&1), half-slot = (k&7) + 8*((k>>4)&1).
__device__ __forceinline__ int amap(int m, int k) {
  return (((m & 15) + (((k >> 3) & 1) << 4)) << 4) + ((k & 7) + (((k >> 4) & 1) << 3));
}

// LDS byte offset of a __shared__ object: flat LDS addresses carry the LDS
// offset in addr[31:0] (aperture in the high half), which is what the async
// VDST operand wants.
__device__ __forceinline__ unsigned ldsOff(const void* p) {
  return (unsigned)(unsigned long long)p;
}

// ---------------------------------------------------------------------------
// Kernel 1: per-node factor GEMMs (collapses the N^2 x CIN einsums)
// ---------------------------------------------------------------------------
__global__ __launch_bounds__(128) void node_precompute_kernel(
    const float* __restrict__ feat, const float* __restrict__ Wmsg0,
    const float* __restrict__ Wmsg1, const float* __restrict__ embed,
    const float* __restrict__ Wmlp1, float* __restrict__ ws)
{
  const int b = blockIdx.x;
  const int t = threadIdx.x;

  if (b == NN) {  // projected position-embedding table PE[33][32]
    float* PE = ws + WS_PE;
    for (int e = t; e < 33 * RBV; e += 128) {
      int r = e >> 5, j = e & 31;
      float s = 0.f;
      for (int k = 0; k < EMBV; k++) s += embed[r * EMBV + k] * Wmlp1[k * RBV + j];
      PE[e] = s;
    }
    return;
  }

  __shared__ float sL[F0V];
  __shared__ float vL[F1V * 3];
  __shared__ float pL[F0V], qL[F0V], v0L[3][F0V];

  for (int i = t; i < F0V; i += 128)     sL[i] = feat[b * FROW + i];
  for (int i = t; i < F1V * 3; i += 128) vL[i] = feat[b * FROW + F0V + i];
  __syncthreads();

  {  // msg0 factors, thread t = output channel o (0..127)
    int o = t;
    float p = Wmsg0[(F0V + F1V) * F0V + o];   // cmf "bias" channel folds into P0
    float q = 0.f;
    float v0[3] = {0.f, 0.f, 0.f};
    for (int c = 0; c < F0V; c++) {
      float s = sL[c];
      p += s * Wmsg0[c * F0V + o];
      q += s * Wmsg0[(F0V + F1V + 1 + c) * F0V + o];
    }
    for (int c = 0; c < F1V; c++) {
      float w = Wmsg0[(F0V + c) * F0V + o];
      v0[0] += vL[c * 3 + 0] * w;
      v0[1] += vL[c * 3 + 1] * w;
      v0[2] += vL[c * 3 + 2] * w;
    }
    ws[WS_P0 + b * F0V + o] = p;
    ws[WS_Q0 + b * F0V + o] = q;
    for (int d = 0; d < 3; d++) ws[WS_V0 + b * 384 + d * F0V + o] = v0[d];
    pL[o] = p; qL[o] = q;
    for (int d = 0; d < 3; d++) v0L[d][o] = v0[d];
  }

  if (t < F1V) {  // msg1 factors
    int o = t;
    float a1[3] = {0,0,0}, b1[3] = {0,0,0}, e1[3] = {0,0,0}, s1 = 0.f;
    for (int c = 0; c < F1V; c++) {
      float wa = Wmsg1[c * F1V + o];
      float wb = Wmsg1[(F1V + F0V + c) * F1V + o];
      float we = Wmsg1[(F1V + F0V + F1V + 1 + c) * F1V + o];
      for (int d = 0; d < 3; d++) {
        float v = vL[c * 3 + d];
        a1[d] += v * wa; b1[d] += v * wb; e1[d] += v * we;
      }
    }
    for (int c = 0; c < F0V; c++) s1 += sL[c] * Wmsg1[(F1V + c) * F1V + o];
    for (int d = 0; d < 3; d++) {
      ws[WS_A1 + b * 192 + d * F1V + o] = a1[d];
      ws[WS_B1 + b * 192 + d * F1V + o] = b1[d];
      ws[WS_E1 + b * 192 + d * F1V + o] = e1[d];
    }
    ws[WS_S1 + b * F1V + o] = s1;
  }
  __syncthreads();

  if (t < RBV) {  // project msg0 factors through W_mlp1[192:320]; Sa through [64:192]
    int j = t;
    float pw = 0.f, qw = 0.f, vw[3] = {0,0,0}, sa = 0.f;
    for (int o = 0; o < F0V; o++) {
      float w = Wmlp1[(EMBV + RBV + F0V + o) * RBV + j];
      pw += pL[o] * w; qw += qL[o] * w;
      vw[0] += v0L[0][o] * w; vw[1] += v0L[1][o] * w; vw[2] += v0L[2][o] * w;
      sa += sL[o] * Wmlp1[(EMBV + RBV + o) * RBV + j];
    }
    ws[WS_P0W + b * RBV + j] = pw;
    ws[WS_Q0W + b * RBV + j] = qw;
    for (int d = 0; d < 3; d++) ws[WS_V0W + b * 96 + d * RBV + j] = vw[d];
    ws[WS_SA + b * RBV + j] = sa;
  }
}

// ---------------------------------------------------------------------------
// Kernel 2: fused pair loop. One wave32 per node a, 2 waves per block.
// ---------------------------------------------------------------------------
__global__ __launch_bounds__(64) void pair_wmma_kernel(
    const float* __restrict__ feat, const float* __restrict__ coord,
    const float* __restrict__ Wmsg1, const float* __restrict__ Wmlp2,
    const float* __restrict__ Wmlp1, const float* __restrict__ Wout0,
    const float* __restrict__ Wout1, const float* __restrict__ Wmove,
    const int* __restrict__ mask, const float* __restrict__ ws,
    float* __restrict__ out)
{
  const float* P0  = ws + WS_P0;  const float* Q0  = ws + WS_Q0;
  const float* V0  = ws + WS_V0;  const float* P0W = ws + WS_P0W;
  const float* Q0W = ws + WS_Q0W; const float* V0W = ws + WS_V0W;
  const float* SA  = ws + WS_SA;  const float* A1  = ws + WS_A1;
  const float* S1  = ws + WS_S1;  const float* B1  = ws + WS_B1;
  const float* E1  = ws + WS_E1;  const float* PE  = ws + WS_PE;

  __shared__ __align__(32) _Float16 wB[12 * 512];     // W_mlp2 B-tiles (32x16 x12)
  __shared__ __align__(32) _Float16 wRad[2 * 512];    // W_mlp1[32:64] B-tiles
  __shared__ __align__(32) _Float16 aTile[2][512];    // per-wave A-tile (16x32 f16)
  __shared__ __align__(16) float gateL[2][16 * NIV];  // per-wave gate tile (f32)
  __shared__ float gCmf[2][16];
  __shared__ float gSh1[2][16][3];
  __shared__ float gG[2][16][9];
  __shared__ int   gRel[2][16];

  const int tid = threadIdx.x;

  // ---- async-DMA raw W_mlp2 (32x192 f32 = 24576B) into the gateL scratch ----
  // GLOBAL_LOAD_ASYNC_TO_LDS_B128: VDST = LDS byte offset, VADDR = 64-bit global
  // address, tracked with ASYNCcnt (no VGPR round trip).
  {
    float* stage = &gateL[0][0];                  // 6144 floats, reused later
    for (int e = tid * 4; e < RBV * NIV; e += 64 * 4) {
      unsigned dst = ldsOff(stage + e);
      const float* srcp = Wmlp2 + e;
      asm volatile("global_load_async_to_lds_b128 %0, %1, off"
                   :: "v"(dst), "v"(srcp) : "memory");
    }
    // Stage rad-weight B-tiles directly (only 2KB): (k,n) -> lane n+16*((k>>3)&1), half (k&7)+8*(k>>4)
    for (int e = tid; e < 2 * 512; e += 64) {
      int t = e >> 9, r = e & 511, k = r >> 4, n = r & 15;
      wRad[(t << 9) + ((n + (((k >> 3) & 1) << 4)) << 4) + ((k & 7) + ((k >> 4) << 3))] =
          (_Float16)Wmlp1[(EMBV + k) * RBV + t * 16 + n];
    }
    asm volatile("s_wait_asynccnt 0x0" ::: "memory");
    __syncthreads();
    // Reformat LDS->LDS into WMMA B-matrix layout, f32 -> f16.
    for (int e = tid; e < 12 * 512; e += 64) {
      int t = e >> 9, r = e & 511, k = r >> 4, n = r & 15;
      wB[(t << 9) + ((n + (((k >> 3) & 1) << 4)) << 4) + ((k & 7) + ((k >> 4) << 3))] =
          (_Float16)stage[k * NIV + t * 16 + n];
    }
  }
  __syncthreads();

  const int w = tid >> 5;      // wave id in block
  const int l = tid & 31;      // lane
  const int a = blockIdx.x * 2 + w;
  const int mA = mask[a];
  const float ca0 = coord[a * 3 + 0], ca1 = coord[a * 3 + 1], ca2 = coord[a * 3 + 2];

  const float inv_cin = rsqrtf((float)CINV);
  const float inv_s3  = 0.57735026919f;     // 1/sqrt(3)
  const float sqrt3   = 1.73205080757f;
  const float inv320  = rsqrtf(320.f);
  const float invRB   = rsqrtf((float)RBV);
  const float invCS   = 1.0f / 0.596f;      // 1/C_SILU (mp-silu constant)
  const float s15 = 3.87298334621f, s5 = 2.2360679775f, s15h = 1.93649167311f;
  const float i10 = 0.31622776601f, i30 = 0.18257418584f;
  const float step = 24.f / 33.f, invstep = 33.f / 24.f;

  float w1da[2];   // W_msg1 "sh1 alone" row, channels o = 2l, 2l+1
  *(float2*)w1da = *(const float2*)(Wmsg1 + (F1V + F0V + F1V) * F1V + 2 * l);

  const int j0 = l & 15, j1 = j0 + 16;
  const int mbase = (l >> 4) << 3;
  const float saj0 = SA[a * RBV + j0], saj1 = SA[a * RBV + j1];

  float acc0[4] = {0, 0, 0, 0};        // owns o = 4l + c   (scalar channels)
  float acc1[2][3] = {{0,0,0},{0,0,0}};// owns o = 2l + c   (vector channels)
  float cntAcc = 0.f;

  for (int chunk = 0; chunk < 16; chunk++) {
    const int m = l & 15;
    const int b = chunk * 16 + m;

    if (chunk < 15) {  // pull next chunk's node factors toward L2/L0
      __builtin_prefetch(&P0[(b + 16) * F0V + 4 * l], 0, 0);
      __builtin_prefetch(&A1[(b + 16) * 192 + 2 * l], 0, 0);
    }

    // ---- geometry (computed redundantly by both half-waves for m = l&15) ----
    const int mB = mask[b];
    const float cm = (mA && mB) ? 1.f : 0.f;
    float vx = (ca0 - coord[b * 3 + 0]) * cm;
    float vy = (ca1 - coord[b * 3 + 1]) * cm;
    float vz = (ca2 - coord[b * 3 + 2]) * cm;
    float nsq = vx * vx + vy * vy + vz * vz;
    float dist = sqrtf(nsq == 0.f ? 1.f : nsq);
    float sh1x = sqrt3 * vx, sh1y = sqrt3 * vy, sh1z = sqrt3 * vz;
    float s2_0 = s15 * vx * vz, s2_1 = s15 * vx * vy;
    float s2_2 = s5 * (vy * vy - 0.5f * (vx * vx + vz * vz));
    float s2_3 = s15 * vy * vz, s2_4 = s15h * (vz * vz - vx * vx);
    int rel = a - b;
    if (rel > KSEQV || rel < -KSEQV) rel = 0;
    if (cm == 0.f) rel = 0;
    rel += KSEQV;

    if (l < 16) {
      gCmf[w][m] = cm;
      gSh1[w][m][0] = sh1x; gSh1[w][m][1] = sh1y; gSh1[w][m][2] = sh1z;
      // G[d][k] = sum_m sh2[m] * M2[m][d][k]  (M2 hard-coded, sparse)
      gG[w][m][0] = -s2_2 * i30 - s2_4 * i10;
      gG[w][m][4] =  2.f * s2_2 * i30;
      gG[w][m][8] = -s2_2 * i30 + s2_4 * i10;
      gG[w][m][1] = s2_1 * i10; gG[w][m][3] = s2_1 * i10;
      gG[w][m][2] = s2_0 * i10; gG[w][m][6] = s2_0 * i10;
      gG[w][m][5] = s2_3 * i10; gG[w][m][7] = s2_3 * i10;
      gRel[w][m] = rel;
      cntAcc += cm;
    }

    // ---- RBF tile straight into A-matrix layout (own contiguous 16 halves) ----
    {
      const int kh = l >> 4;
      for (int h = 0; h < 16; h++) {
        int k = (h & 7) + (kh << 3) + ((h >> 3) << 4);
        float t0 = (dist - (float)(k + 1) * step) * invstep;
        aTile[w][l * 16 + h] = (_Float16)(__expf(-t0 * t0) * (1.f / 1.12f) * cm);
      }
    }
    __syncthreads();

    // ---- rad @ W_mlp1[32:64] via WMMA (M=16 b's, K=32 RBF, N=32) ----
    v16h av = *(const v16h*)&aTile[w][l * 16];
    v16h br0 = *(const v16h*)&wRad[l * 16];
    v16h br1 = *(const v16h*)&wRad[512 + l * 16];
    v8f c0 = {}, c1 = {};
    c0 = __builtin_amdgcn_wmma_f32_16x16x32_f16(false, av, false, br0, (short)0, c0, false, false);
    c1 = __builtin_amdgcn_wmma_f32_16x16x32_f16(false, av, false, br1, (short)0, c1, false, false);
    __syncthreads();  // rad tile consumed; safe to overwrite with h

    // ---- add factored extras, nsilu, rebuild A-tile as h ----
    for (int v = 0; v < 8; v++) {
      int mm = mbase + v;
      int bb = chunk * 16 + mm;
      float cmm = gCmf[w][mm];
      const float* sh = gSh1[w][mm];
      int rr = gRel[w][mm];
      float e0 = PE[rr * RBV + j0] + saj0 +
                 (cmm * P0W[bb * RBV + j0] + Q0W[bb * RBV + j0] +
                  inv_s3 * (sh[0] * V0W[bb * 96 + j0] + sh[1] * V0W[bb * 96 + 32 + j0] +
                            sh[2] * V0W[bb * 96 + 64 + j0])) * inv_cin;
      float e1v = PE[rr * RBV + j1] + saj1 +
                 (cmm * P0W[bb * RBV + j1] + Q0W[bb * RBV + j1] +
                  inv_s3 * (sh[0] * V0W[bb * 96 + j1] + sh[1] * V0W[bb * 96 + 32 + j1] +
                            sh[2] * V0W[bb * 96 + 64 + j1])) * inv_cin;
      aTile[w][amap(mm, j0)] = (_Float16)(siluf((c0[v] + e0) * inv320) * invCS);
      aTile[w][amap(mm, j1)] = (_Float16)(siluf((c1[v] + e1v) * inv320) * invCS);
    }
    __syncthreads();

    // ---- gate = h @ W_mlp2 / sqrt(32): 12 WMMAs (N-tiles of 16) ----
    v16h hv = *(const v16h*)&aTile[w][l * 16];
    for (int t = 0; t < 12; t++) {
      v16h bv = *(const v16h*)&wB[(t << 9) + l * 16];
      v8f cg = {};
      cg = __builtin_amdgcn_wmma_f32_16x16x32_f16(false, hv, false, bv, (short)0, cg, false, false);
      for (int v = 0; v < 8; v++)
        gateL[w][(mbase + v) * NIV + t * 16 + j0] = cg[v] * invRB;
    }
    __syncthreads();

    // ---- aggregation: rebuild msg0/msg1 from node factors (vectorized) ----
    for (int mi = 0; mi < 16; mi++) {
      float cm2 = gCmf[w][mi];
      if (cm2 == 0.f) continue;  // wave-uniform branch
      int bb = chunk * 16 + mi;
      const float* sh = gSh1[w][mi];
      const float* GG = gG[w][mi];

      // scalar channels, o = 4l + c : all b128 loads
      float p0a[4], q0a[4], vxa[4], vya[4], vza[4], g0a[4];
      *(float4*)p0a = *(const float4*)(P0 + bb * F0V + 4 * l);
      *(float4*)q0a = *(const float4*)(Q0 + bb * F0V + 4 * l);
      *(float4*)vxa = *(const float4*)(V0 + bb * 384 + 4 * l);
      *(float4*)vya = *(const float4*)(V0 + bb * 384 + 128 + 4 * l);
      *(float4*)vza = *(const float4*)(V0 + bb * 384 + 256 + 4 * l);
      *(float4*)g0a = *(const float4*)(gateL[w] + mi * NIV + 4 * l);
      for (int c = 0; c < 4; c++) {
        float msg = cm2 * p0a[c] + q0a[c] +
                    inv_s3 * (sh[0] * vxa[c] + sh[1] * vya[c] + sh[2] * vza[c]);
        acc0[c] += msg * inv_cin * g0a[c] * cm2;
      }

      // vector channels, o = 2l + c : b64 loads
      float ga[2], s1a[2], a1a[3][2], b1a[3][2], e1a[3][2];
      *(float2*)ga  = *(const float2*)(gateL[w] + mi * NIV + F0V + 2 * l);
      *(float2*)s1a = *(const float2*)(S1 + bb * F1V + 2 * l);
      for (int d = 0; d < 3; d++) {
        *(float2*)a1a[d] = *(const float2*)(A1 + bb * 192 + d * 64 + 2 * l);
        *(float2*)b1a[d] = *(const float2*)(B1 + bb * 192 + d * 64 + 2 * l);
        *(float2*)e1a[d] = *(const float2*)(E1 + bb * 192 + d * 64 + 2 * l);
      }
      for (int c = 0; c < 2; c++) {
        float g = ga[c] * cm2;
        float s1w = s1a[c] + w1da[c];
        for (int d = 0; d < 3; d++) {
          float m1 = cm2 * a1a[d][c] + sh[d] * s1w +
                     sqrt3 * (b1a[0][c] * GG[d] + b1a[1][c] * GG[3 + d] + b1a[2][c] * GG[6 + d]) +
                     e1a[d][c];
          acc1[c][d] += m1 * inv_cin * g;
        }
      }
    }
    __syncthreads();  // protect shared tiles before next chunk
  }

  // ---- normalize aggregates ----
  float cnt = cntAcc;
  for (int off = 16; off; off >>= 1) cnt += __shfl_xor(cnt, off, 32);
  float sc = (1.f / (cnt + 1e-6f)) * ((cnt > 1.f) ? 1.f : 0.f);

  {
    float tmp[4];
    for (int c = 0; c < 4; c++) tmp[c] = acc0[c] * sc;
    *(float4*)(gateL[w] + 4 * l) = *(float4*)tmp;     // AG0[o], o = 4l+c
  }
  for (int c = 0; c < 2; c++)
    for (int d = 0; d < 3; d++)
      gateL[w][F0V + (2 * l + c) * 3 + d] = acc1[c][d] * sc;  // AG1[o][d]
  __syncthreads();

  const float* AG0 = &gateL[w][0];
  const float* AG1 = &gateL[w][F0V];
  const float inv_f0 = rsqrtf((float)F0V), inv_f1 = rsqrtf((float)F1V);

  // scalar head: f0, layernorm
  float r0v[4], muP = 0.f;
  for (int c = 0; c < 4; c++) {
    int o = c * 32 + l;
    float f = 0.f;
    for (int cc = 0; cc < F0V; cc++) f += AG0[cc] * Wout0[cc * F0V + o];
    float r = feat[a * FROW + o] + f * inv_f0;
    r0v[c] = r; muP += r;
  }
  for (int off = 16; off; off >>= 1) muP += __shfl_xor(muP, off, 32);
  float mu = muP / (float)F0V;
  float vP = 0.f;
  for (int c = 0; c < 4; c++) { float d = r0v[c] - mu; vP += d * d; }
  for (int off = 16; off; off >>= 1) vP += __shfl_xor(vP, off, 32);
  float rstd = rsqrtf(vP / (float)F0V + 1e-6f);
  for (int c = 0; c < 4; c++) {
    int o = c * 32 + l;
    out[a * OUTD + o] = (r0v[c] - mu) * rstd;
  }

  // vector head: f1, RMS norm, coord update
  float r1v[2][3], ssP = 0.f, mvP[3] = {0, 0, 0};
  for (int c = 0; c < 2; c++) {
    int o = c * 32 + l;
    for (int d = 0; d < 3; d++) {
      float f = 0.f;
      for (int cc = 0; cc < F1V; cc++) f += AG1[cc * 3 + d] * Wout1[cc * F1V + o];
      f *= inv_f1;
      float r = feat[a * FROW + F0V + o * 3 + d] + f;
      r1v[c][d] = r; ssP += r * r;
      mvP[d] += f * Wmove[o];
    }
  }
  for (int off = 16; off; off >>= 1) {
    ssP += __shfl_xor(ssP, off, 32);
    for (int d = 0; d < 3; d++) mvP[d] += __shfl_xor(mvP[d], off, 32);
  }
  float rinv = rsqrtf(ssP / (float)NIV + 1e-6f);
  for (int c = 0; c < 2; c++) {
    int o = c * 32 + l;
    for (int d = 0; d < 3; d++)
      out[a * OUTD + F0V + o * 3 + d] = r1v[c][d] * rinv;
  }
  if (l == 0)
    for (int d = 0; d < 3; d++)
      out[a * OUTD + FROW + d] = coord[a * 3 + d] + 0.001f * mvP[d] * inv_f1;
}

extern "C" void kernel_launch(void* const* d_in, const int* in_sizes, int n_in,
                              void* d_out, int out_size, void* d_ws, size_t ws_size,
                              hipStream_t stream) {
  (void)in_sizes; (void)n_in; (void)out_size; (void)ws_size;
  const float* feat  = (const float*)d_in[0];
  const float* coord = (const float*)d_in[1];
  const float* Wmsg0 = (const float*)d_in[2];
  const float* Wmsg1 = (const float*)d_in[3];
  const float* embed = (const float*)d_in[4];
  const float* Wmlp1 = (const float*)d_in[5];
  const float* Wmlp2 = (const float*)d_in[6];
  const float* Wout0 = (const float*)d_in[7];
  const float* Wout1 = (const float*)d_in[8];
  const float* Wmove = (const float*)d_in[9];
  const int*   maskp = (const int*)d_in[10];
  float* ws  = (float*)d_ws;
  float* out = (float*)d_out;

  node_precompute_kernel<<<NN + 1, 128, 0, stream>>>(feat, Wmsg0, Wmsg1, embed, Wmlp1, ws);
  pair_wmma_kernel<<<NN / 2, 64, 0, stream>>>(feat, coord, Wmsg1, Wmlp2, Wmlp1,
                                              Wout0, Wout1, Wmove, maskp, ws, out);
}
